// AlphaQubitLikeModel_19576460935857
// MI455X (gfx1250) — compile-verified, matrix operand order checked by reading, not run on hardware
//
#include <hip/hip_runtime.h>
#include <hip/hip_bf16.h>

// ---------------------------------------------------------------------------
// MI455X (gfx1250): one workgroup (8 wave32) per batch element, all GEMMs via
// v_wmma_f32_16x16x32_f16. Each wave owns an N-tile and sweeps all M-tiles with
// multiple accumulators so every B fragment (global) is loaded exactly once.
// ---------------------------------------------------------------------------

typedef __attribute__((ext_vector_type(16))) _Float16 v16h;
typedef __attribute__((ext_vector_type(8)))  _Float16 v8h;
typedef __attribute__((ext_vector_type(8)))  float    v8f;

constexpr int SEQ   = 24;     // L
constexpr int DIM   = 256;    // D
constexpr int NHEAD = 4;
constexpr int HD    = 64;     // DA == DM
constexpr int DBIAS = 16;
constexpr int RD    = 48;

constexpr int SXN_LD  = 264;  // f16 row stride (halves) for 256-wide panels
constexpr int SVT_LD  = 40;   // V^T / probs row stride (halves)
constexpr int SGLU_LD = 520;  // glu activations stride (halves)

__device__ __forceinline__ float gelu_f(float x) {
  return 0.5f * x * (1.0f + erff(x * 0.70710678118654752f));
}

union FragU { struct { v8h lo, hi; } p; v16h v; };

// CDNA5 16-bit 16x32 fragment layout:
// lanes 0-15 hold rows m (K 0..7 / 16..23), lanes 16-31 same rows (K 8..15 / 24..31).
__device__ __forceinline__ v16h ld_frag(const _Float16* base, int ld, int rbase,
                                        int k0, int lane) {
  const _Float16* p = base + (rbase + (lane & 15)) * ld + k0 + (lane >> 4) * 8;
  FragU u;
  u.p.lo = *(const v8h*)p;
  u.p.hi = *(const v8h*)(p + 16);
  return u.v;
}

__device__ __forceinline__ v16h ld_frag_row(const _Float16* rowptr, int lane) {
  const _Float16* p = rowptr + (lane >> 4) * 8;
  FragU u;
  u.p.lo = *(const v8h*)p;
  u.p.hi = *(const v8h*)(p + 16);
  return u.v;
}

__device__ __forceinline__ v8f wmma16(v16h a, v16h b, v8f c) {
  return __builtin_amdgcn_wmma_f32_16x16x32_f16(false, a, false, b, (short)0, c,
                                                false, false);
}

// LayerNorm: one wave per row, shfl_xor reductions, writes f16 panel.
__device__ void ln_rows_f16(const float* src, int srcld, _Float16* dst, int dstld,
                            int nrows, const float* g, const float* bt,
                            int wave, int lane) {
  for (int r = wave; r < nrows; r += 8) {
    const float* x = src + r * srcld;
    float s = 0.f, s2 = 0.f;
    for (int c = lane; c < DIM; c += 32) { float v = x[c]; s += v; s2 += v * v; }
#pragma unroll
    for (int off = 16; off >= 1; off >>= 1) {
      s  += __shfl_xor(s,  off, 32);
      s2 += __shfl_xor(s2, off, 32);
    }
    float mean = s * (1.f / DIM);
    float rstd = rsqrtf(s2 * (1.f / DIM) - mean * mean + 1e-5f);
    _Float16* d = dst + r * dstld;
    for (int c = lane; c < DIM; c += 32)
      d[c] = (_Float16)((x[c] - mean) * rstd * g[c] + bt[c]);
  }
}

struct LayerW {
  const float *ln1_g, *ln1_b, *ln2_g, *ln2_b;
  const float *Wb, *bq, *bk, *bv, *bo, *fc1b, *fc2b;
  const _Float16 *wq, *wk, *wv, *wo, *wfc1, *wfc2;
};

struct NetW {
  const int *syndrome, *stab_id, *cycle_id;
  const float *bias;
  const float *stab_emb, *cycle_emb, *val_emb;
  LayerW lyr[3];
  const float *convP;
  const float *cln_g[2], *cln_b[2], *cbias[2];
  const _Float16 *wconv[2];
  const float *fln_g, *fln_b;
  const float *rP, *scb, *drb, *outw, *outb;
  const _Float16 *wsc, *wdr;
  const float *bw1[16], *bb1[16], *bw2[16], *bb2[16];
  float *out;
};

// ---------------------------------------------------------------------------
// weight prep: f32 -> f16 in WMMA-friendly [N][K] panels
// ---------------------------------------------------------------------------
__global__ void prep_transpose(const float* in, _Float16* out, int K, int N) {
  int idx = blockIdx.x * 256 + threadIdx.x;
  if (idx < K * N) { int k = idx / N, n = idx % N; out[n * K + k] = (_Float16)in[idx]; }
}
__global__ void prep_qkv(const float* in, _Float16* out) {  // (4,256,64) -> [h*64+e][k]
  int idx = blockIdx.x * 256 + threadIdx.x;
  if (idx < 65536) {
    int h = idx >> 14, k = (idx >> 6) & 255, e = idx & 63;
    out[(h * 64 + e) * 256 + k] = (_Float16)in[idx];
  }
}
__global__ void prep_conv(const float* in, _Float16* out, int T) { // [co][ci][t]->[t][co][ci]
  int idx = blockIdx.x * 256 + threadIdx.x;
  int total = 256 * 256 * T;
  if (idx < total) {
    int co = idx / (256 * T);
    int r  = idx % (256 * T);
    int ci = r / T, t = r % T;
    out[(t * 256 + co) * 256 + ci] = (_Float16)in[idx];
  }
}
__global__ void prep_cvt(const float* in, _Float16* out, int n) {
  int idx = blockIdx.x * 256 + threadIdx.x;
  if (idx < n) out[idx] = (_Float16)in[idx];
}

// ---------------------------------------------------------------------------
// main decoder kernel: one block == one batch element
// ---------------------------------------------------------------------------
__global__ __launch_bounds__(256) void aq_decoder(NetW P) {
  __shared__ float     sX[32 * DIM];           // residual stream (rows 24..31 = 0)
  __shared__ _Float16  sXn[32 * SXN_LD];       // LN output, WMMA A operand
  __shared__ __align__(16) char sReg[103424];  // phase-overlapped region

  const int tid = threadIdx.x, wave = tid >> 5, lane = tid & 31;
  const int b = blockIdx.x;
  const int lm = lane & 15, lhi = lane >> 4;

  // attention-phase views
  _Float16* sQ  = (_Float16*)(sReg);            // 32x264 f16
  _Float16* sK  = (_Float16*)(sReg + 16896);    // 32x264 f16
  _Float16* sVT = (_Float16*)(sReg + 33792);    // 256x40 f16 (V transposed)
  _Float16* sO  = (_Float16*)(sReg + 54272);    // 32x264 f16 (attn out, concat heads)
  float*    sSc = (float*)   (sReg + 71168);    // 4x24x26 f32 raw scores
  float*    sBp = (float*)   (sReg + 81152);    // 4x24x24 f32 pairwise bias
  _Float16* sPr = (_Float16*)(sReg + 90368);    // 4x32x40 f16 probs (A operand)
  // ffn-phase view
  _Float16* sGlu = (_Float16*)(sReg);           // 32x520 f16
  // conv-phase views
  float*    sG  = (float*)   (sReg);            // 36x256 f32 grid
  _Float16* sGn = (_Float16*)(sReg + 36864);    // 48x264 f16 (rows 36..47 zero)
  // readout-phase views
  _Float16* sG16 = (_Float16*)(sReg);           // 48x264 f16 scattered grid
  _Float16* sR9  = (_Float16*)(sReg + 25344);   // 16x264 f16 (9 valid rows)
  float*    sR2  = (float*)   (sReg + 33792);   // 9x48 f32
  float*    sV1  = (float*)   (sReg + 35584);   // 48 f32
  float*    sV2  = (float*)   (sReg + 35776);   // 48 f32

  // ---- embeddings: X = (stab + cycle + val)/sqrt(2); pad rows zeroed
  for (int i = tid; i < 32 * DIM; i += 256) {
    int r = i >> 8, c = i & 255;
    float v = 0.f;
    if (r < SEQ) {
      int sid = P.stab_id[r], cid = P.cycle_id[r], sy = P.syndrome[b * SEQ + r];
      v = (P.stab_emb[sid * DIM + c] + P.cycle_emb[cid * DIM + c] +
           P.val_emb[sy * DIM + c]) * 0.70710678118654752f;
    }
    sX[i] = v;
  }
  for (int i = tid; i < 32 * SXN_LD; i += 256) sXn[i] = (_Float16)0.f;
  __syncthreads();

#pragma unroll
  for (int l = 0; l < 3; ++l) {
    const LayerW& W = P.lyr[l];

    // ======== attention ========
    ln_rows_f16(sX, DIM, sXn, SXN_LD, SEQ, W.ln1_g, W.ln1_b, wave, lane);
    for (int i = tid; i < NHEAD * 32 * SVT_LD; i += 256) sPr[i] = (_Float16)0.f;
    // warm L2 for the big conv weight panels used later this layer
    for (int i = tid; i < 1152; i += 256) {
      __builtin_prefetch(P.wconv[0] + (size_t)i * 1024, 0, 1);
      __builtin_prefetch(P.wconv[1] + (size_t)i * 1024, 0, 1);
    }
    for (int ij = tid; ij < SEQ * SEQ; ij += 256) {
      int i = ij / SEQ, j = ij % SEQ;
      const float* bp = P.bias + (((size_t)b * SEQ + i) * SEQ + j) * DBIAS;
      float h0 = 0, h1 = 0, h2 = 0, h3 = 0;
#pragma unroll
      for (int d = 0; d < DBIAS; ++d) {
        float v = bp[d];
        h0 += v * W.Wb[d * 4 + 0]; h1 += v * W.Wb[d * 4 + 1];
        h2 += v * W.Wb[d * 4 + 2]; h3 += v * W.Wb[d * 4 + 3];
      }
      sBp[(0 * SEQ + i) * SEQ + j] = h0; sBp[(1 * SEQ + i) * SEQ + j] = h1;
      sBp[(2 * SEQ + i) * SEQ + j] = h2; sBp[(3 * SEQ + i) * SEQ + j] = h3;
    }
    __syncthreads();

    // Q/K/V projections: 48 N-tile tasks, each sweeps both M-tiles (proj is
    // wave-uniform -> scalar branches only).
    for (int t = wave; t < 48; t += 8) {
      int proj = t >> 4, nt = t & 15;
      const _Float16* wB  = proj == 0 ? W.wq : (proj == 1 ? W.wk : W.wv);
      const float*    bia = proj == 0 ? W.bq : (proj == 1 ? W.bk : W.bv);
      v8f acc0 = {}, acc1 = {};
      for (int k0 = 0; k0 < DIM; k0 += 32) {
        v16h bb = ld_frag(wB, DIM, nt * 16, k0, lane);
        v16h a0 = ld_frag(sXn, SXN_LD, 0,  k0, lane);
        v16h a1 = ld_frag(sXn, SXN_LD, 16, k0, lane);
        acc0 = wmma16(a0, bb, acc0);
        acc1 = wmma16(a1, bb, acc1);
      }
      int n = nt * 16 + lm;
      float bval = bia[(n >> 6) * 64 + (n & 63)];
      if (proj == 2) {
#pragma unroll
        for (int v = 0; v < 8; ++v) {
          int m0 = lhi * 8 + v;
          sVT[n * SVT_LD + m0]      = (_Float16)(acc0[v] + bval);
          sVT[n * SVT_LD + 16 + m0] = (_Float16)(acc1[v] + bval);
        }
      } else {
        _Float16* dst = proj == 0 ? sQ : sK;
#pragma unroll
        for (int v = 0; v < 8; ++v) {
          int m0 = lhi * 8 + v;
          dst[m0 * SXN_LD + n]        = (_Float16)(acc0[v] + bval);
          dst[(16 + m0) * SXN_LD + n] = (_Float16)(acc1[v] + bval);
        }
      }
    }
    __syncthreads();

    // S = Q K^T per head: 8 tasks (h, nt), both M-tiles share the B fragment
    for (int t = wave; t < 8; t += 8) {
      int h = t >> 1, nt = t & 1;
      v8f acc0 = {}, acc1 = {};
#pragma unroll
      for (int kk = 0; kk < HD; kk += 32) {
        v16h bb = ld_frag(sK, SXN_LD, nt * 16, h * 64 + kk, lane);
        v16h a0 = ld_frag(sQ, SXN_LD, 0,  h * 64 + kk, lane);
        v16h a1 = ld_frag(sQ, SXN_LD, 16, h * 64 + kk, lane);
        acc0 = wmma16(a0, bb, acc0);
        acc1 = wmma16(a1, bb, acc1);
      }
      int n = nt * 16 + lm;
      if (n < SEQ) {
#pragma unroll
        for (int v = 0; v < 8; ++v) {
          int m0 = lhi * 8 + v;
          sSc[(h * SEQ + m0) * 26 + n] = acc0[v];
          int m1 = 16 + m0;
          if (m1 < SEQ) sSc[(h * SEQ + m1) * 26 + n] = acc1[v];
        }
      }
    }
    __syncthreads();

    // softmax((S + Bp)/sqrt(64)) -> f16 prob panel (zero-padded to 32x32)
    for (int t = tid; t < NHEAD * SEQ; t += 256) {
      int h = t / SEQ, i = t % SEQ;
      float lg[SEQ];
      float mx = -1e30f;
#pragma unroll
      for (int j = 0; j < SEQ; ++j) {
        float v = (sSc[(h * SEQ + i) * 26 + j] + sBp[(h * SEQ + i) * SEQ + j]) * 0.125f;
        lg[j] = v; mx = fmaxf(mx, v);
      }
      float sum = 0.f;
#pragma unroll
      for (int j = 0; j < SEQ; ++j) { lg[j] = __expf(lg[j] - mx); sum += lg[j]; }
      float inv = 1.f / sum;
#pragma unroll
      for (int j = 0; j < SEQ; ++j)
        sPr[(h * 32 + i) * SVT_LD + j] = (_Float16)(lg[j] * inv);
    }
    __syncthreads();

    // O = P @ V: 16 tasks (h, nt-of-4); K=32 (padded tokens); shared B fragment
    for (int t = wave; t < 16; t += 8) {
      int h = t >> 2, nt = t & 3;
      v16h bb = ld_frag(sVT, SVT_LD, h * 64 + nt * 16, 0, lane);
      v16h a0 = ld_frag(sPr + h * 32 * SVT_LD, SVT_LD, 0,  0, lane);
      v16h a1 = ld_frag(sPr + h * 32 * SVT_LD, SVT_LD, 16, 0, lane);
      v8f acc0 = {}, acc1 = {};
      acc0 = wmma16(a0, bb, acc0);
      acc1 = wmma16(a1, bb, acc1);
      int n = h * 64 + nt * 16 + lm;
#pragma unroll
      for (int v = 0; v < 8; ++v) {
        int m0 = lhi * 8 + v;
        sO[m0 * SXN_LD + n]        = (_Float16)acc0[v];  // rows >= 24 are exactly 0
        sO[(16 + m0) * SXN_LD + n] = (_Float16)acc1[v];
      }
    }
    __syncthreads();

    // Wo projection + residual: 16 N-tile tasks, shared B fragments
    for (int nt = wave; nt < 16; nt += 8) {
      v8f acc0 = {}, acc1 = {};
      for (int k0 = 0; k0 < DIM; k0 += 32) {
        v16h bb = ld_frag(W.wo, DIM, nt * 16, k0, lane);
        v16h a0 = ld_frag(sO, SXN_LD, 0,  k0, lane);
        v16h a1 = ld_frag(sO, SXN_LD, 16, k0, lane);
        acc0 = wmma16(a0, bb, acc0);
        acc1 = wmma16(a1, bb, acc1);
      }
      int n = nt * 16 + lm;
      float bo_ = W.bo[n];
#pragma unroll
      for (int v = 0; v < 8; ++v) {
        int m0 = lhi * 8 + v;
        sX[m0 * DIM + n] += acc0[v] + bo_;
        int m1 = 16 + m0;
        if (m1 < SEQ) sX[m1 * DIM + n] += acc1[v] + bo_;
      }
    }
    __syncthreads();

    // ======== GLU FFN ========
    ln_rows_f16(sX, DIM, sXn, SXN_LD, SEQ, W.ln2_g, W.ln2_b, wave, lane);
    __syncthreads();
    for (int jt = wave; jt < 32; jt += 8) {   // 32 glu-col tiles; 4 accumulators
      v8f aA0 = {}, aG0 = {}, aA1 = {}, aG1 = {};
      for (int k0 = 0; k0 < DIM; k0 += 32) {
        v16h bA = ld_frag(W.wfc1, DIM, jt * 16, k0, lane);
        v16h bG = ld_frag(W.wfc1, DIM, 512 + jt * 16, k0, lane);
        v16h a0 = ld_frag(sXn, SXN_LD, 0,  k0, lane);
        v16h a1 = ld_frag(sXn, SXN_LD, 16, k0, lane);
        aA0 = wmma16(a0, bA, aA0);
        aG0 = wmma16(a0, bG, aG0);
        aA1 = wmma16(a1, bA, aA1);
        aG1 = wmma16(a1, bG, aG1);
      }
      int n = jt * 16 + lm;
      float ba = W.fc1b[n], bg = W.fc1b[512 + n];
#pragma unroll
      for (int v = 0; v < 8; ++v) {
        int m0 = lhi * 8 + v;
        sGlu[m0 * SGLU_LD + n] =
            (_Float16)(gelu_f(aA0[v] + ba) * (aG0[v] + bg));
        sGlu[(16 + m0) * SGLU_LD + n] =
            (_Float16)(gelu_f(aA1[v] + ba) * (aG1[v] + bg));
      }
    }
    __syncthreads();
    for (int nt = wave; nt < 16; nt += 8) {   // fc2, K=512, shared B fragments
      v8f acc0 = {}, acc1 = {};
      for (int k0 = 0; k0 < 512; k0 += 32) {
        v16h bb = ld_frag(W.wfc2, 512, nt * 16, k0, lane);
        v16h a0 = ld_frag(sGlu, SGLU_LD, 0,  k0, lane);
        v16h a1 = ld_frag(sGlu, SGLU_LD, 16, k0, lane);
        acc0 = wmma16(a0, bb, acc0);
        acc1 = wmma16(a1, bb, acc1);
      }
      int n = nt * 16 + lm;
      float bo_ = W.fc2b[n];
#pragma unroll
      for (int v = 0; v < 8; ++v) {
        int m0 = lhi * 8 + v;
        sX[m0 * DIM + n] += acc0[v] + bo_;
        int m1 = 16 + m0;
        if (m1 < SEQ) sX[m1 * DIM + n] += acc1[v] + bo_;
      }
    }
    __syncthreads();

    // ======== conv block on 6x6 grid (shared weights) ========
    for (int i = tid; i < 36 * DIM; i += 256) {
      int p = i >> 8, c = i & 255;
      sG[i] = (p < SEQ) ? sX[p * DIM + c] : P.convP[c];
    }
    __syncthreads();
#pragma unroll
    for (int cl = 0; cl < 2; ++cl) {
      for (int i = tid; i < 12 * SXN_LD; i += 256) sGn[36 * SXN_LD + i] = (_Float16)0.f;
      ln_rows_f16(sG, DIM, sGn, SXN_LD, 36, P.cln_g[cl], P.cln_b[cl], wave, lane);
      __syncthreads();
      const _Float16* wc = P.wconv[cl];
      const float*    cb = P.cbias[cl];
      // 16 N-tile tasks; 3 M-tiles (48 rows) share each B fragment.
      for (int nt = wave; nt < 16; nt += 8) {
        v8f acc0 = {}, acc1 = {}, acc2 = {};
        int p0 = lm, p1 = 16 + lm, p2 = 32 + lm;
#pragma unroll
        for (int tap = 0; tap < 9; ++tap) {   // implicit GEMM: 9 taps x K=256
          int dy = tap / 3 - 1, dx = tap % 3 - 1;
          auto halo = [&](int p) -> int {
            int qy = p / 6 + dy, qx = p % 6 + dx;
            bool ok = (p < 36) && (qy >= 0) && (qy < 6) && (qx >= 0) && (qx < 6);
            return ok ? (qy * 6 + qx) : 47;   // row 47 is all zeros
          };
          const _Float16* r0 = sGn + halo(p0) * SXN_LD;
          const _Float16* r1 = sGn + halo(p1) * SXN_LD;
          const _Float16* r2 = sGn + halo(p2) * SXN_LD;
          const _Float16* wt = wc + tap * (256 * 256);
          for (int k0 = 0; k0 < DIM; k0 += 32) {
            v16h bb = ld_frag(wt, DIM, nt * 16, k0, lane);
            acc0 = wmma16(ld_frag_row(r0 + k0, lane), bb, acc0);
            acc1 = wmma16(ld_frag_row(r1 + k0, lane), bb, acc1);
            acc2 = wmma16(ld_frag_row(r2 + k0, lane), bb, acc2);
          }
        }
        int n = nt * 16 + lm;
        float cbv = cb[n];
#pragma unroll
        for (int v = 0; v < 8; ++v) {
          int m0 = lhi * 8 + v;
          sG[m0 * DIM + n]        += gelu_f(acc0[v] + cbv);
          sG[(16 + m0) * DIM + n] += gelu_f(acc1[v] + cbv);
          int m2 = 32 + m0;
          if (m2 < 36) sG[m2 * DIM + n] += gelu_f(acc2[v] + cbv);
        }
      }
      __syncthreads();
    }
    for (int i = tid; i < SEQ * DIM; i += 256) sX[i] = sG[i];  // gather tokens 0..23
    __syncthreads();
  }

  // ======== final LN + readout ========
  ln_rows_f16(sX, DIM, sXn, SXN_LD, SEQ, P.fln_g, P.fln_b, wave, lane);
  __syncthreads();
  for (int i = tid; i < 48 * SXN_LD; i += 256) {
    int p = i / SXN_LD, c = i % SXN_LD;
    _Float16 v = (_Float16)0.f;
    if (c < DIM) {
      if (p < SEQ)      v = sXn[p * SXN_LD + c];
      else if (p < 36)  v = (_Float16)P.rP[c];
    }
    sG16[i] = v;
  }
  for (int i = tid; i < 16 * SXN_LD; i += 256) sR9[i] = (_Float16)0.f;
  __syncthreads();

  // 2x2 stride-2 conv (6x6 -> 3x3), implicit GEMM: 4 taps x K=256
  for (int nt = wave; nt < 16; nt += 8) {
    int p = lm;
    int oy = p / 3, ox = p % 3;
    v8f acc = {};
#pragma unroll
    for (int tap = 0; tap < 4; ++tap) {
      int qy = 2 * oy + (tap >> 1), qx = 2 * ox + (tap & 1);
      int qrow = (p < 9) ? (qy * 6 + qx) : 47;
      const _Float16* arow = sG16 + qrow * SXN_LD;
      const _Float16* wt   = P.wsc + tap * (256 * 256);
      for (int k0 = 0; k0 < DIM; k0 += 32) {
        v16h a  = ld_frag_row(arow + k0, lane);
        v16h bb = ld_frag(wt, DIM, nt * 16, k0, lane);
        acc = wmma16(a, bb, acc);
      }
    }
    int n = nt * 16 + lm;
    float sb = P.scb[n];
#pragma unroll
    for (int v = 0; v < 8; ++v) {
      int m = lhi * 8 + v;
      if (m < 9) sR9[m * SXN_LD + n] = (_Float16)gelu_f(acc[v] + sb);
    }
  }
  __syncthreads();

  // 1x1 conv D->48, then gelu
  for (int t = wave; t < 3; t += 8) {
    v8f acc = {};
    for (int k0 = 0; k0 < DIM; k0 += 32) {
      v16h a  = ld_frag(sR9, SXN_LD, 0, k0, lane);
      v16h bb = ld_frag(P.wdr, DIM, t * 16, k0, lane);
      acc = wmma16(a, bb, acc);
    }
    int n = t * 16 + lm;
    float db = P.drb[n];
#pragma unroll
    for (int v = 0; v < 8; ++v) {
      int m = lhi * 8 + v;
      if (m < 9) sR2[m * RD + n] = gelu_f(acc[v] + db);
    }
  }
  __syncthreads();

  if (tid < RD) {                       // spatial mean over 3x3
    float s = 0.f;
    for (int m = 0; m < 9; ++m) s += sR2[m * RD + tid];
    sV1[tid] = s * (1.f / 9.f);
  }
  __syncthreads();

  for (int blk = 0; blk < 16; ++blk) {  // tiny 48-wide residual MLPs (VALU)
    const float *w1 = P.bw1[blk], *b1 = P.bb1[blk];
    const float *w2 = P.bw2[blk], *b2 = P.bb2[blk];
    float h = 0.f;
    if (tid < RD) {
      float s = b1[tid];
      for (int k = 0; k < RD; ++k) s += sV1[k] * w1[k * RD + tid];
      h = gelu_f(s);
    }
    __syncthreads();
    if (tid < RD) sV2[tid] = h;
    __syncthreads();
    if (tid < RD) {
      float s = b2[tid];
      for (int k = 0; k < RD; ++k) s += sV2[k] * w2[k * RD + tid];
      sV1[tid] += s;
    }
    __syncthreads();
  }
  if (tid == 0) {
    float s = P.outb[0];
    for (int k = 0; k < RD; ++k) s += sV1[k] * P.outw[k];
    P.out[b] = s;
  }
}

// ---------------------------------------------------------------------------
// host-side launch
// ---------------------------------------------------------------------------
extern "C" void kernel_launch(void* const* d_in, const int* in_sizes, int n_in,
                              void* d_out, int out_size, void* d_ws, size_t ws_size,
                              hipStream_t stream) {
  (void)n_in; (void)ws_size;
  // Flattening assumed: top-level dict order (syndrome, stab_id, cycle_id, bias,
  // params) with params flattened as a jax pytree (dict keys sorted, lists/tuples
  // in order):
  //   params: conv, cycle_emb, final_ln_b, final_ln_g, layers, readout,
  //           stab_emb, val_emb
  //   conv: P, convs[(w,b)x2], lns[(g,b)x2]
  //   layer: Wb, Wk, Wo, Wq, Wv, bk, bo, bq, bv, fc1_b, fc1_w, fc2_b, fc2_w,
  //          ln1_b, ln1_g, ln2_b, ln2_g
  //   readout: P, blocks[(b1,b2,w1,w2)x16], dr_b, dr_w, out_b, out_w, sc_b, sc_w
  auto F = [&](int i) { return (const float*)d_in[i]; };

  NetW P;
  P.syndrome = (const int*)d_in[0];
  P.stab_id  = (const int*)d_in[1];
  P.cycle_id = (const int*)d_in[2];
  P.bias     = F(3);
  const int base = 4;
  P.convP = F(base + 0);
  const float* conv_w[2] = { F(base + 1), F(base + 3) };
  P.cbias[0] = F(base + 2); P.cbias[1] = F(base + 4);
  P.cln_g[0] = F(base + 5); P.cln_b[0] = F(base + 6);
  P.cln_g[1] = F(base + 7); P.cln_b[1] = F(base + 8);
  P.cycle_emb = F(base + 9);
  P.fln_b = F(base + 10);
  P.fln_g = F(base + 11);

  const int li = base + 12;             // layers: 17 tensors each (sorted keys)
  const float *Wq_f[3], *Wk_f[3], *Wv_f[3], *Wo_f[3], *fc1_f[3], *fc2_f[3];
  for (int l = 0; l < 3; ++l) {
    int o = li + 17 * l;
    P.lyr[l].Wb   = F(o + 0);
    Wk_f[l]       = F(o + 1);
    Wo_f[l]       = F(o + 2);
    Wq_f[l]       = F(o + 3);
    Wv_f[l]       = F(o + 4);
    P.lyr[l].bk   = F(o + 5);
    P.lyr[l].bo   = F(o + 6);
    P.lyr[l].bq   = F(o + 7);
    P.lyr[l].bv   = F(o + 8);
    P.lyr[l].fc1b = F(o + 9);
    fc1_f[l]      = F(o + 10);
    P.lyr[l].fc2b = F(o + 11);
    fc2_f[l]      = F(o + 12);
    P.lyr[l].ln1_b = F(o + 13);
    P.lyr[l].ln1_g = F(o + 14);
    P.lyr[l].ln2_b = F(o + 15);
    P.lyr[l].ln2_g = F(o + 16);
  }
  const int ro = li + 51;               // readout
  P.rP = F(ro + 0);
  for (int k = 0; k < 16; ++k) {
    int o = ro + 1 + 4 * k;
    P.bb1[k] = F(o + 0);
    P.bb2[k] = F(o + 1);
    P.bw1[k] = F(o + 2);
    P.bw2[k] = F(o + 3);
  }
  P.drb = F(ro + 65);
  const float* dr_w = F(ro + 66);
  P.outb = F(ro + 67);
  P.outw = F(ro + 68);
  P.scb  = F(ro + 69);
  const float* sc_w = F(ro + 70);
  P.stab_emb = F(ro + 71);
  P.val_emb  = F(ro + 72);
  P.out = (float*)d_out;

  // ---- carve f16 weight panels out of d_ws and run prep conversions
  _Float16* cur = (_Float16*)d_ws;
  auto take = [&](size_t halves) { _Float16* p = cur; cur += halves; return p; };
  for (int l = 0; l < 3; ++l) {
    _Float16* wq = take(65536);  _Float16* wk = take(65536);
    _Float16* wv = take(65536);  _Float16* wo = take(65536);
    _Float16* w1 = take(262144); _Float16* w2 = take(131072);
    P.lyr[l].wq = wq; P.lyr[l].wk = wk; P.lyr[l].wv = wv; P.lyr[l].wo = wo;
    P.lyr[l].wfc1 = w1; P.lyr[l].wfc2 = w2;
    prep_qkv<<<256, 256, 0, stream>>>(Wq_f[l], wq);
    prep_qkv<<<256, 256, 0, stream>>>(Wk_f[l], wk);
    prep_qkv<<<256, 256, 0, stream>>>(Wv_f[l], wv);
    prep_transpose<<<(65536 + 255) / 256, 256, 0, stream>>>(Wo_f[l], wo, 256, 256);
    prep_transpose<<<(262144 + 255) / 256, 256, 0, stream>>>(fc1_f[l], w1, 256, 1024);
    prep_transpose<<<(131072 + 255) / 256, 256, 0, stream>>>(fc2_f[l], w2, 512, 256);
  }
  for (int c = 0; c < 2; ++c) {
    _Float16* wc = take(589824);
    P.wconv[c] = wc;
    prep_conv<<<(589824 + 255) / 256, 256, 0, stream>>>(conv_w[c], wc, 9);
  }
  {
    _Float16* wsc = take(262144);
    P.wsc = wsc;
    prep_conv<<<(262144 + 255) / 256, 256, 0, stream>>>(sc_w, wsc, 4);
    _Float16* wdr = take(12288);
    P.wdr = wdr;
    prep_cvt<<<(12288 + 255) / 256, 256, 0, stream>>>(dr_w, wdr, 12288);
  }

  const int B = in_sizes[0] / SEQ;      // 2048
  (void)out_size;
  aq_decoder<<<B, 256, 0, stream>>>(P);
}
